// ChebBase_26010321944990
// MI455X (gfx1250) — matrix operand order repaired and checked by reading, not compiled, not run on hardware
//
#include <hip/hip_runtime.h>
#include <hip/hip_bf16.h>

typedef __attribute__((ext_vector_type(16))) __bf16 v16bf;
typedef __attribute__((ext_vector_type(8)))  float  v8f;

#define F_IN 512
#define HDIM 128
#define CDIM 40
#define CQ   (CDIM / 4)   // 10 float4 quads per row
#define NT1  8            // 128 / 16 output col tiles for GEMM1
#define NT2  3            // 48 / 16 output col tiles for GEMM2 (40 padded to 48)
#define KT1  (F_IN / 32)  // 16 k-tiles for GEMM1
#define KT2  (HDIM / 32)  // 4 k-tiles for GEMM2

// ---------- helpers ----------
static __device__ __forceinline__ __bf16 f2bf(float f) {
    return (__bf16)f;     // native fptrunc -> v_cvt_pk_bf16_f32
}

// Load one A operand (16x32 bf16 tile slice for this lane) from an f32 row.
// ISA A-layout (16-bit, 16x32): lane<16 holds M=lane, elems 0..7 = K khalf*8..+7,
// elems 8..15 = K 16+khalf*8..+7 (khalf = lane>>4).
static __device__ __forceinline__ v16bf load_a_f32(const float* __restrict__ xr, int base1) {
    float4 q0 = *(const float4*)(xr + base1);
    float4 q1 = *(const float4*)(xr + base1 + 4);
    float4 q2 = *(const float4*)(xr + base1 + 16);
    float4 q3 = *(const float4*)(xr + base1 + 20);
    v16bf a;
    a[0]  = f2bf(q0.x); a[1]  = f2bf(q0.y); a[2]  = f2bf(q0.z); a[3]  = f2bf(q0.w);
    a[4]  = f2bf(q1.x); a[5]  = f2bf(q1.y); a[6]  = f2bf(q1.z); a[7]  = f2bf(q1.w);
    a[8]  = f2bf(q2.x); a[9]  = f2bf(q2.y); a[10] = f2bf(q2.z); a[11] = f2bf(q2.w);
    a[12] = f2bf(q3.x); a[13] = f2bf(q3.y); a[14] = f2bf(q3.z); a[15] = f2bf(q3.w);
    return a;
}

// ---------- weight repack: f32 [K,Ncols] -> bf16 B-tile layout ----------
// Tile t=(kt,nt): lane l (n=l&15, khalf=l>>4) holds 16 contiguous bf16,
// elem e -> W[kt*32 + khalf*8 + (e&7) + (e>>3)*16][nt*16 + n].  One 32B load per lane later.
__global__ void k_conv_w1(const float* __restrict__ W1, __bf16* __restrict__ W1bf) {
    int idx = blockIdx.x * blockDim.x + threadIdx.x;
    if (idx >= KT1 * NT1 * 32 * 16) return;
    int e = idx & 15, lane = (idx >> 4) & 31, t = idx >> 9;
    int kt = t >> 3, nt = t & 7;
    int n = lane & 15, khalf = lane >> 4;
    int K = kt * 32 + khalf * 8 + (e & 7) + ((e >> 3) << 4);
    int c = nt * 16 + n;
    W1bf[idx] = f2bf(W1[K * HDIM + c]);
}

__global__ void k_conv_w2(const float* __restrict__ W2, __bf16* __restrict__ W2bf) {
    int idx = blockIdx.x * blockDim.x + threadIdx.x;
    if (idx >= KT2 * NT2 * 32 * 16) return;
    int e = idx & 15, lane = (idx >> 4) & 31, t = idx >> 9;
    int kt = t / NT2, nt = t % NT2;
    int n = lane & 15, khalf = lane >> 4;
    int K = kt * 32 + khalf * 8 + (e & 7) + ((e >> 3) << 4);
    int c = nt * 16 + n;
    W2bf[idx] = (c < CDIM) ? f2bf(W2[K * CDIM + c]) : f2bf(0.0f);
}

// ---------- GEMM1: h = relu(x @ W1 + b1), per-wave 16x128 tile ----------
// Software-pipelined: B tile t+1 and A tile kt+1 load while WMMA t executes.
__global__ void __launch_bounds__(256) k_gemm1(const float* __restrict__ x,
                                               const __bf16* __restrict__ W1bf,
                                               const float* __restrict__ b1,
                                               float* __restrict__ h,
                                               int N, int ntiles) {
    int lane = threadIdx.x & 31, wave = threadIdx.x >> 5;
    int tile = blockIdx.x * 8 + wave;
    if (tile >= ntiles) return;                  // wave-uniform: EXEC stays all-1s
    int rowbase = tile * 16;
    int m = lane & 15, khalf = lane >> 4;
    int arow = rowbase + m; if (arow >= N) arow = N - 1;
    const float* xr = x + (size_t)arow * F_IN;
    const v16bf* bp = (const v16bf*)W1bf;

    v8f accv[NT1];
#pragma unroll
    for (int i = 0; i < NT1; ++i) accv[i] = (v8f){0,0,0,0,0,0,0,0};

    v16bf a = load_a_f32(xr, khalf * 8);         // kt = 0
    v16bf b = bp[lane];                          // tile 0
#pragma unroll
    for (int kt = 0; kt < KT1; ++kt) {
        // prefetch next k-tile's A while this k-tile's 8 WMMAs run
        v16bf an = (kt < KT1 - 1) ? load_a_f32(xr, (kt + 1) * 32 + khalf * 8) : a;
#pragma unroll
        for (int nt = 0; nt < NT1; ++nt) {
            int tnext = kt * NT1 + nt + 1;
            if (tnext >= KT1 * NT1) tnext = KT1 * NT1 - 1;
            v16bf bn = bp[tnext * 32 + lane];    // prefetch next B tile
            accv[nt] = __builtin_amdgcn_wmma_f32_16x16x32_bf16(
                false, a, false, b, (short)0, accv[nt], false, false);
            b = bn;
        }
        a = an;
    }
    // C/D layout: VGPR j, lane<16 -> M=j col=lane ; lane>=16 -> M=8+j col=lane-16
    int coln = lane & 15, roff = khalf * 8;
#pragma unroll
    for (int nt = 0; nt < NT1; ++nt) {
        int c = nt * 16 + coln;
        float bias = b1[c];
#pragma unroll
        for (int j = 0; j < 8; ++j) {
            int r = rowbase + roff + j;
            if (r < N) {
                float v = accv[nt][j] + bias;
                h[(size_t)r * HDIM + c] = v > 0.f ? v : 0.f;
            }
        }
    }
}

// ---------- GEMM2: t0 = h @ W2 + b2, per-wave 16x48 tile (cols>=40 masked) ----------
__global__ void __launch_bounds__(256) k_gemm2(const float* __restrict__ h,
                                               const __bf16* __restrict__ W2bf,
                                               const float* __restrict__ b2,
                                               float* __restrict__ t0,
                                               int N, int ntiles) {
    int lane = threadIdx.x & 31, wave = threadIdx.x >> 5;
    int tile = blockIdx.x * 8 + wave;
    if (tile >= ntiles) return;
    int rowbase = tile * 16;
    int m = lane & 15, khalf = lane >> 4;
    int arow = rowbase + m; if (arow >= N) arow = N - 1;
    const float* xr = h + (size_t)arow * HDIM;
    const v16bf* bp = (const v16bf*)W2bf;

    v8f accv[NT2];
#pragma unroll
    for (int i = 0; i < NT2; ++i) accv[i] = (v8f){0,0,0,0,0,0,0,0};

    v16bf a = load_a_f32(xr, khalf * 8);
    v16bf b = bp[lane];
#pragma unroll
    for (int kt = 0; kt < KT2; ++kt) {
        v16bf an = (kt < KT2 - 1) ? load_a_f32(xr, (kt + 1) * 32 + khalf * 8) : a;
#pragma unroll
        for (int nt = 0; nt < NT2; ++nt) {
            int tnext = kt * NT2 + nt + 1;
            if (tnext >= KT2 * NT2) tnext = KT2 * NT2 - 1;
            v16bf bn = bp[tnext * 32 + lane];
            accv[nt] = __builtin_amdgcn_wmma_f32_16x16x32_bf16(
                false, a, false, b, (short)0, accv[nt], false, false);
            b = bn;
        }
        a = an;
    }
    int coln = lane & 15, roff = khalf * 8;
#pragma unroll
    for (int nt = 0; nt < NT2; ++nt) {
        int c = nt * 16 + coln;
        if (c < CDIM) {
            float bias = b2[c];
#pragma unroll
            for (int j = 0; j < 8; ++j) {
                int r = rowbase + roff + j;
                if (r < N) t0[(size_t)r * CDIM + c] = accv[nt][j] + bias;
            }
        }
    }
}

// ---------- degree / normalization ----------
__global__ void k_fill1(float* __restrict__ deg, int N) {
    int i = blockIdx.x * blockDim.x + threadIdx.x;
    if (i < N) deg[i] = 1.0f;                 // self-loop contribution
}
__global__ void k_deg(const int* __restrict__ row, float* __restrict__ deg, int E) {
    int e = blockIdx.x * blockDim.x + threadIdx.x;
    if (e < E)
        __hip_atomic_fetch_add(&deg[row[e]], 1.0f, __ATOMIC_RELAXED, __HIP_MEMORY_SCOPE_AGENT);
}
__global__ void k_rsqrt(float* __restrict__ dis, int N) {
    int i = blockIdx.x * blockDim.x + threadIdx.x;
    if (i < N) dis[i] = rsqrtf(dis[i]);       // deg >= 1 always (self loops)
}

// ---------- prop(v): y = A_hat @ v ----------
// self-loop term: y[i,:] = dis[i]^2 * v[i,:]   (one thread per float4 quad)
__global__ void k_prop_self(const float* __restrict__ dis, const float* __restrict__ v,
                            float* __restrict__ y, int total /* N*CQ */) {
    int idx = blockIdx.x * blockDim.x + threadIdx.x;
    if (idx >= total) return;
    int i = idx / CQ;
    int q = idx - i * CQ;
    float d = dis[i]; d *= d;
    const float4 vv = *(const float4*)(v + (size_t)i * CDIM + q * 4);
    *(float4*)(y + (size_t)i * CDIM + q * 4) =
        make_float4(d * vv.x, d * vv.y, d * vv.z, d * vv.w);
}
// edge scatter: one thread per (edge, quad); b128 gathers, shared index/dis loads
__global__ void k_prop_edges(const int* __restrict__ row, const int* __restrict__ col,
                             const float* __restrict__ dis, const float* __restrict__ v,
                             float* __restrict__ y, int total /* E*CQ */) {
    int idx = blockIdx.x * blockDim.x + threadIdx.x;
    if (idx >= total) return;
    int e = idx / CQ;
    int q = idx - e * CQ;
    int r = row[e], cl = col[e];
    float w = dis[r] * dis[cl];
    const float4 vv = *(const float4*)(v + (size_t)cl * CDIM + q * 4);
    float* yp = y + (size_t)r * CDIM + q * 4;
    __hip_atomic_fetch_add(yp + 0, w * vv.x, __ATOMIC_RELAXED, __HIP_MEMORY_SCOPE_AGENT);
    __hip_atomic_fetch_add(yp + 1, w * vv.y, __ATOMIC_RELAXED, __HIP_MEMORY_SCOPE_AGENT);
    __hip_atomic_fetch_add(yp + 2, w * vv.z, __ATOMIC_RELAXED, __HIP_MEMORY_SCOPE_AGENT);
    __hip_atomic_fetch_add(yp + 3, w * vv.w, __ATOMIC_RELAXED, __HIP_MEMORY_SCOPE_AGENT);
}

// ---------- Chebyshev elementwise (float4) ----------
__global__ void k_acc_init(const float* __restrict__ t0, const float* __restrict__ t1,
                           const float* __restrict__ temp, float* __restrict__ acc,
                           int total4 /* N*CDIM/4 */) {
    int idx = blockIdx.x * blockDim.x + threadIdx.x;
    if (idx >= total4) return;
    float c0 = temp[0], c1 = temp[1];
    float4 a = ((const float4*)t0)[idx];
    float4 b = ((const float4*)t1)[idx];
    ((float4*)acc)[idx] = make_float4(c0 * a.x + c1 * b.x, c0 * a.y + c1 * b.y,
                                      c0 * a.z + c1 * b.z, c0 * a.w + c1 * b.w);
}
// y := 2*y - t0  (becomes Tx_k), acc += temp[k]*y
__global__ void k_cheb(float* __restrict__ y, const float* __restrict__ t0,
                       float* __restrict__ acc, const float* __restrict__ temp,
                       int k, int total4) {
    int idx = blockIdx.x * blockDim.x + threadIdx.x;
    if (idx >= total4) return;
    float coe = temp[k];
    float4 yy = ((const float4*)y)[idx];
    float4 tt = ((const float4*)t0)[idx];
    float4 t2 = make_float4(2.f * yy.x - tt.x, 2.f * yy.y - tt.y,
                            2.f * yy.z - tt.z, 2.f * yy.w - tt.w);
    ((float4*)y)[idx] = t2;
    float4 av = ((const float4*)acc)[idx];
    ((float4*)acc)[idx] = make_float4(av.x + coe * t2.x, av.y + coe * t2.y,
                                      av.z + coe * t2.z, av.w + coe * t2.w);
}

// ---------- log_softmax over C=40, one thread per row (float4 loads) ----------
__global__ void k_lsm(const float* __restrict__ a, float* __restrict__ out, int N) {
    int i = blockIdx.x * blockDim.x + threadIdx.x;
    if (i >= N) return;
    const float4* r = (const float4*)(a + (size_t)i * CDIM);
    float4 rv[CQ];
#pragma unroll
    for (int q = 0; q < CQ; ++q) rv[q] = r[q];
    float mx = rv[0].x;
#pragma unroll
    for (int q = 0; q < CQ; ++q) {
        mx = fmaxf(mx, rv[q].x); mx = fmaxf(mx, rv[q].y);
        mx = fmaxf(mx, rv[q].z); mx = fmaxf(mx, rv[q].w);
    }
    float s = 0.f;
#pragma unroll
    for (int q = 0; q < CQ; ++q) {
        s += __expf(rv[q].x - mx); s += __expf(rv[q].y - mx);
        s += __expf(rv[q].z - mx); s += __expf(rv[q].w - mx);
    }
    float lse = mx + __logf(s);
    float4* o = (float4*)(out + (size_t)i * CDIM);
#pragma unroll
    for (int q = 0; q < CQ; ++q)
        o[q] = make_float4(rv[q].x - lse, rv[q].y - lse, rv[q].z - lse, rv[q].w - lse);
}

// ---------- host ----------
extern "C" void kernel_launch(void* const* d_in, const int* in_sizes, int n_in,
                              void* d_out, int out_size, void* d_ws, size_t ws_size,
                              hipStream_t stream) {
    (void)n_in; (void)out_size; (void)ws_size;
    const float* x    = (const float*)d_in[0];
    const int*   ei   = (const int*)d_in[1];
    const float* W1   = (const float*)d_in[2];
    const float* b1   = (const float*)d_in[3];
    const float* W2   = (const float*)d_in[4];
    const float* b2   = (const float*)d_in[5];
    const float* temp = (const float*)d_in[6];
    float* out = (float*)d_out;

    const int N = in_sizes[0] / F_IN;
    const int E = in_sizes[1] / 2;
    const int Kp1 = in_sizes[6];          // K+1 coefficients
    const int* row = ei;
    const int* col = ei + E;

    char* ws = (char*)d_ws;
    size_t off = 0;
    auto alloc = [&](size_t bytes) -> void* {
        void* p = ws + off;
        off += (bytes + 255) & ~(size_t)255;
        return p;
    };
    __bf16* W1bf = (__bf16*)alloc((size_t)KT1 * NT1 * 32 * 16 * 2);
    __bf16* W2bf = (__bf16*)alloc((size_t)KT2 * NT2 * 32 * 16 * 2);
    float* h   = (float*)alloc((size_t)N * HDIM * 4);
    float* t0  = (float*)alloc((size_t)N * CDIM * 4);
    float* t1  = (float*)alloc((size_t)N * CDIM * 4);
    float* yb  = (float*)alloc((size_t)N * CDIM * 4);
    float* acc = (float*)alloc((size_t)N * CDIM * 4);
    float* dis = (float*)alloc((size_t)N * 4);

    const int ntiles = (N + 15) / 16;
    const int gblk   = (ntiles + 7) / 8;     // 8 waves per 256-thread block
    const int ncq = N * CQ;                  // quads per node array
    const int ecq = E * CQ;                  // quads over edges
    const int nc4 = (N * CDIM) / 4;          // float4 count for elementwise

    k_conv_w1<<<(KT1 * NT1 * 32 * 16 + 255) / 256, 256, 0, stream>>>(W1, W1bf);
    k_conv_w2<<<(KT2 * NT2 * 32 * 16 + 255) / 256, 256, 0, stream>>>(W2, W2bf);
    k_gemm1<<<gblk, 256, 0, stream>>>(x, W1bf, b1, h, N, ntiles);
    k_gemm2<<<gblk, 256, 0, stream>>>(h, W2bf, b2, t0, N, ntiles);

    k_fill1<<<(N + 255) / 256, 256, 0, stream>>>(dis, N);
    k_deg<<<(E + 255) / 256, 256, 0, stream>>>(row, dis, E);
    k_rsqrt<<<(N + 255) / 256, 256, 0, stream>>>(dis, N);

    // Tx1 = prop(Tx0)
    k_prop_self<<<(ncq + 255) / 256, 256, 0, stream>>>(dis, t0, t1, ncq);
    k_prop_edges<<<(ecq + 255) / 256, 256, 0, stream>>>(row, col, dis, t0, t1, ecq);
    k_acc_init<<<(nc4 + 255) / 256, 256, 0, stream>>>(t0, t1, temp, acc, nc4);

    float* c0 = t0; float* c1 = t1; float* cy = yb;
    for (int k = 2; k < Kp1; ++k) {
        k_prop_self<<<(ncq + 255) / 256, 256, 0, stream>>>(dis, c1, cy, ncq);
        k_prop_edges<<<(ecq + 255) / 256, 256, 0, stream>>>(row, col, dis, c1, cy, ecq);
        k_cheb<<<(nc4 + 255) / 256, 256, 0, stream>>>(cy, c0, acc, temp, k, nc4);
        float* t = c0; c0 = c1; c1 = cy; cy = t;
    }
    k_lsm<<<(N + 255) / 256, 256, 0, stream>>>(acc, out, N);
}
